// WeightedMMDLoss_81080392613939
// MI455X (gfx1250) — matrix operand order repaired and checked by reading, not compiled, not run on hardware
//
#include <hip/hip_runtime.h>
#include <hip/hip_bf16.h>
#include <math.h>

// ---------------------------------------------------------------------------
// Weighted MMD loss, fused RBF-kernel GEMMs on CDNA5 WMMA (bf16 hi/lo split),
// async-to-LDS double-buffered pipeline (gfx1250 ASYNCcnt path).
//   xx = w K(N,N) w ; yy = u K(R,R) u ; xy = w K(N,R) u ; out = sqrt(xx+yy-2xy)
//   w_m w_n K(m,n) = ea[m]*eb[n]*exp(2*gamma*(a_m . b_n))
// ---------------------------------------------------------------------------

typedef __attribute__((ext_vector_type(16))) __bf16 v16bf;
typedef __attribute__((ext_vector_type(8)))  __bf16 bf16x8;
typedef __attribute__((ext_vector_type(4)))  __bf16 bf16x4;
typedef __attribute__((ext_vector_type(8)))  float  v8f;
typedef __attribute__((ext_vector_type(4)))  float  f32x4;

#define MMD_NROWS 8192
#define MMD_DIMS  512
#define MMD_GAMMA 0.001f
#define MMD_BLK   128
#define MMD_KC    32
#define MMD_KPAD  40                       // LDS row stride in bf16 elems (16B-aligned, bank-skewed)
#define MMD_KSTEPS (MMD_DIMS / MMD_KC)     // 16
#define MMD_TILES  (MMD_NROWS / MMD_BLK)   // 64
#define MMD_PLANE_ELEMS ((size_t)MMD_NROWS * MMD_DIMS)

__device__ __forceinline__ __bf16 mmd_bf_round(float x) {
    unsigned u = __builtin_bit_cast(unsigned, x);
    u += 0x7FFFu + ((u >> 16) & 1u);          // round-to-nearest-even
    unsigned short s = (unsigned short)(u >> 16);
    return __builtin_bit_cast(__bf16, s);
}
__device__ __forceinline__ float mmd_bf_to_f32(__bf16 h) {
    unsigned u = ((unsigned)__builtin_bit_cast(unsigned short, h)) << 16;
    return __builtin_bit_cast(float, u);
}

// ---------------------------------------------------------------------------
// Shared WMMA K-step: 8 accumulator tiles per wave (2 row x 4 col of 16x16),
// fp32 emulated as hi*hi + hi*lo + lo*hi bf16 WMMAs.
// ---------------------------------------------------------------------------
__device__ __forceinline__ void mmd_compute_step(const __bf16* sAhi, const __bf16* sAlo,
                                                 const __bf16* sBhi, const __bf16* sBlo,
                                                 int lane, int wm, int wn, v8f* acc) {
    // A fragments (ISA 16-bit A 16x32): lanes 0-15 K{0-7,16-23}, lanes 16-31 K{8-15,24-31}
    v16bf aHi[2], aLo[2];
    const int ar = lane & 15;
    const int ka = (lane < 16) ? 0 : 8;
#pragma unroll
    for (int i = 0; i < 2; ++i) {
        int base = (wm + i * 16 + ar) * MMD_KPAD + ka;
        union { bf16x8 h[2]; v16bf v; } u;
        u.h[0] = *(const bf16x8*)&sAhi[base];
        u.h[1] = *(const bf16x8*)&sAhi[base + 16];
        aHi[i] = u.v;
        u.h[0] = *(const bf16x8*)&sAlo[base];
        u.h[1] = *(const bf16x8*)&sAlo[base + 16];
        aLo[i] = u.v;
    }
    // B fragments (32x16): lane = column, lanes 0-15 K{0-7,16-23}, lanes 16-31 K{8-15,24-31}
    const int kb = (lane < 16) ? 0 : 8;
#pragma unroll
    for (int j = 0; j < 4; ++j) {
        int base = (wn + j * 16 + ar) * MMD_KPAD + kb;
        union { bf16x8 h[2]; v16bf v; } u;
        u.h[0] = *(const bf16x8*)&sBhi[base];
        u.h[1] = *(const bf16x8*)&sBhi[base + 16];
        v16bf bHi = u.v;
        u.h[0] = *(const bf16x8*)&sBlo[base];
        u.h[1] = *(const bf16x8*)&sBlo[base + 16];
        v16bf bLo = u.v;
#pragma unroll
        for (int i = 0; i < 2; ++i) {
            int a = i * 4 + j;
            acc[a] = __builtin_amdgcn_wmma_f32_16x16x32_bf16(false, aHi[i], false, bHi,
                                                             (short)0, acc[a], false, false);
            acc[a] = __builtin_amdgcn_wmma_f32_16x16x32_bf16(false, aHi[i], false, bLo,
                                                             (short)0, acc[a], false, false);
            acc[a] = __builtin_amdgcn_wmma_f32_16x16x32_bf16(false, aLo[i], false, bHi,
                                                             (short)0, acc[a], false, false);
        }
    }
}

// Epilogue: psum = sum_mn ea[m]*eb[n]*exp(2g*acc).  C/D layout: VGPR r -> M=r(+8 hi lanes), N=lane&15
__device__ __forceinline__ float mmd_epilogue(const v8f* acc, const float* ea, const float* eb,
                                              int bm, int bn, int wm, int wn, int lane) {
    const float c2 = 2.0f * MMD_GAMMA;
    float psum = 0.0f;
#pragma unroll
    for (int i = 0; i < 2; ++i) {
        const int mbase = bm + wm + i * 16 + ((lane >> 4) << 3);
        float eam[8];
#pragma unroll
        for (int r = 0; r < 8; ++r) eam[r] = ea[mbase + r];
#pragma unroll
        for (int j = 0; j < 4; ++j) {
            const int n = bn + wn + j * 16 + (lane & 15);
            const float ebn = eb[n];
            v8f a = acc[i * 4 + j];
            float s = 0.0f;
#pragma unroll
            for (int r = 0; r < 8; ++r) s += eam[r] * __expf(c2 * a[r]);
            psum += ebn * s;
        }
    }
    return psum;
}

// ---------------------------------------------------------------------------
// Prep: ea[m] = w[m]*exp(-g*||N_m||^2), eb[m] = (1/8192)*exp(-g*||R_m||^2)
// ---------------------------------------------------------------------------
__global__ void mmd_prep_kernel(const float* __restrict__ Nmat,
                                const float* __restrict__ Rmat,
                                const float* __restrict__ w,
                                float* __restrict__ eNw,
                                float* __restrict__ eRu) {
    __shared__ float red[128];
    const int row = blockIdx.x;
    const int t   = threadIdx.x;
    const float* n = Nmat + (size_t)row * MMD_DIMS;
    const float* r = Rmat + (size_t)row * MMD_DIMS;
    float sn = 0.f, sr = 0.f;
    for (int k = t; k < MMD_DIMS; k += 128) {
        float a = n[k]; sn += a * a;
        float b = r[k]; sr += b * b;
    }
    red[t] = sn; __syncthreads();
    for (int off = 64; off > 0; off >>= 1) { if (t < off) red[t] += red[t + off]; __syncthreads(); }
    float normN = red[0]; __syncthreads();
    red[t] = sr; __syncthreads();
    for (int off = 64; off > 0; off >>= 1) { if (t < off) red[t] += red[t + off]; __syncthreads(); }
    float normR = red[0];
    if (t == 0) {
        eNw[row] = w[row] * __expf(-MMD_GAMMA * normN);
        eRu[row] = (1.0f / (float)MMD_NROWS) * __expf(-MMD_GAMMA * normR);
    }
}

// ---------------------------------------------------------------------------
// One-pass f32 -> bf16 hi/lo plane conversion (bandwidth bound, done once)
// ---------------------------------------------------------------------------
__global__ void mmd_convert_kernel(const float* __restrict__ Nmat, const float* __restrict__ Rmat,
                                   unsigned short* __restrict__ NhiU, unsigned short* __restrict__ NloU,
                                   unsigned short* __restrict__ RhiU, unsigned short* __restrict__ RloU) {
    __bf16* Nhi = (__bf16*)NhiU; __bf16* Nlo = (__bf16*)NloU;
    __bf16* Rhi = (__bf16*)RhiU; __bf16* Rlo = (__bf16*)RloU;
    size_t idx = ((size_t)blockIdx.x * 256 + threadIdx.x) * 4;
    f32x4 a = *(const f32x4*)(Nmat + idx);
    f32x4 b = *(const f32x4*)(Rmat + idx);
    bf16x4 ahi, alo, bhi, blo;
#pragma unroll
    for (int e = 0; e < 4; ++e) {
        __bf16 h = mmd_bf_round(a[e]);
        ahi[e] = h; alo[e] = mmd_bf_round(a[e] - mmd_bf_to_f32(h));
        h = mmd_bf_round(b[e]);
        bhi[e] = h; blo[e] = mmd_bf_round(b[e] - mmd_bf_to_f32(h));
    }
    *(bf16x4*)(Nhi + idx) = ahi;
    *(bf16x4*)(Nlo + idx) = alo;
    *(bf16x4*)(Rhi + idx) = bhi;
    *(bf16x4*)(Rlo + idx) = blo;
}

// ---------------------------------------------------------------------------
// Fast path: pre-converted bf16 planes, async-to-LDS double-buffered GEMM.
// blockIdx.z selects term: 0=xx 1=yy 2=xy
// ---------------------------------------------------------------------------
__global__ __launch_bounds__(256)
void mmd_wmma_async_kernel(const unsigned short* __restrict__ NhiU, const unsigned short* __restrict__ NloU,
                           const unsigned short* __restrict__ RhiU, const unsigned short* __restrict__ RloU,
                           const float* __restrict__ eNw, const float* __restrict__ eRu,
                           float* __restrict__ partials) {
    __shared__ __attribute__((aligned(16))) __bf16 smem[2][4][MMD_BLK * MMD_KPAD];
    __shared__ float red[256];

    const int term = blockIdx.z;
    const __bf16* Nhi = (const __bf16*)NhiU; const __bf16* Nlo = (const __bf16*)NloU;
    const __bf16* Rhi = (const __bf16*)RhiU; const __bf16* Rlo = (const __bf16*)RloU;
    const __bf16* pl[4];
    pl[0] = (term == 1) ? Rhi : Nhi;   // A hi
    pl[1] = (term == 1) ? Rlo : Nlo;   // A lo
    pl[2] = (term == 0) ? Nhi : Rhi;   // B hi
    pl[3] = (term == 0) ? Nlo : Rlo;   // B lo
    const float* ea = (term == 1) ? eRu : eNw;
    const float* eb = (term == 0) ? eNw : eRu;

    const int bm = blockIdx.x * MMD_BLK;
    const int bn = blockIdx.y * MMD_BLK;
    const int t    = threadIdx.x;
    const int lane = t & 31;
    const int wid  = t >> 5;
    const int wm   = (wid & 3) * 32;
    const int wn   = (wid >> 2) * 64;

    // Issue 8 async B128 transfers (one 128x32 bf16 tile = 8KB per plane, 4 planes)
    auto stage = [&](int buf, int ks) {
#pragma unroll
        for (int p = 0; p < 4; ++p) {
            const int rbase = (p < 2) ? bm : bn;
#pragma unroll
            for (int i = 0; i < 2; ++i) {
                int idx = t + i * 256;           // 0..511
                int row = idx >> 2;              // 0..127
                int sub = idx & 3;               // 8 bf16 (16B) chunk within the 32-K row
                unsigned lds = (unsigned)(size_t)&smem[buf][p][row * MMD_KPAD + sub * 8];
                unsigned long long gaddr =
                    (unsigned long long)(size_t)(pl[p] + ((size_t)(rbase + row) * MMD_DIMS
                                                          + ks * MMD_KC + sub * 8));
                asm volatile("global_load_async_to_lds_b128 %0, %1, off"
                             :: "v"(lds), "v"(gaddr) : "memory");
            }
        }
    };

    v8f zero = {};
    v8f acc[8];
#pragma unroll
    for (int i = 0; i < 8; ++i) acc[i] = zero;

    stage(0, 0);
    for (int ks = 0; ks < MMD_KSTEPS; ++ks) {
        const int cur = ks & 1;
        if (ks + 1 < MMD_KSTEPS) {
            stage(cur ^ 1, ks + 1);                       // prefetch next slab into other buffer
            asm volatile("s_wait_asynccnt 0x8" ::: "memory");  // oldest 8 (current buf) complete
        } else {
            asm volatile("s_wait_asynccnt 0x0" ::: "memory");
        }
        __syncthreads();   // all waves' current buffer transfers complete
        mmd_compute_step(smem[cur][0], smem[cur][1], smem[cur][2], smem[cur][3],
                         lane, wm, wn, acc);
        __syncthreads();   // all waves done reading buf[cur] before it is restaged
    }

    red[t] = mmd_epilogue(acc, ea, eb, bm, bn, wm, wn, lane);
    __syncthreads();
    for (int off = 128; off > 0; off >>= 1) { if (t < off) red[t] += red[t + off]; __syncthreads(); }
    if (t == 0)
        partials[(size_t)term * (MMD_TILES * MMD_TILES) + blockIdx.y * MMD_TILES + blockIdx.x] = red[0];
}

// ---------------------------------------------------------------------------
// Fallback path (workspace too small for planes): convert f32->bf16 in-kernel.
// ---------------------------------------------------------------------------
__global__ __launch_bounds__(256)
void mmd_wmma_fb_kernel(const float* __restrict__ Nmat, const float* __restrict__ Rmat,
                        const float* __restrict__ eNw, const float* __restrict__ eRu,
                        float* __restrict__ partials) {
    __shared__ __attribute__((aligned(16))) __bf16 sAhi[MMD_BLK * MMD_KPAD];
    __shared__ __attribute__((aligned(16))) __bf16 sAlo[MMD_BLK * MMD_KPAD];
    __shared__ __attribute__((aligned(16))) __bf16 sBhi[MMD_BLK * MMD_KPAD];
    __shared__ __attribute__((aligned(16))) __bf16 sBlo[MMD_BLK * MMD_KPAD];
    __shared__ float red[256];

    const int term = blockIdx.z;
    const float* A  = (term == 1) ? Rmat : Nmat;
    const float* B  = (term == 0) ? Nmat : Rmat;
    const float* ea = (term == 1) ? eRu  : eNw;
    const float* eb = (term == 0) ? eNw  : eRu;

    const int bm = blockIdx.x * MMD_BLK;
    const int bn = blockIdx.y * MMD_BLK;
    const int t    = threadIdx.x;
    const int lane = t & 31;
    const int wid  = t >> 5;
    const int wm   = (wid & 3) * 32;
    const int wn   = (wid >> 2) * 64;

    v8f zero = {};
    v8f acc[8];
#pragma unroll
    for (int i = 0; i < 8; ++i) acc[i] = zero;

    for (int ks = 0; ks < MMD_KSTEPS; ++ks) {
        __syncthreads();
#pragma unroll
        for (int i = 0; i < 4; ++i) {
            int f   = t + i * 256;
            int row = f >> 3;
            int k4  = (f & 7) << 2;
            f32x4 va = *(const f32x4*)(A + (size_t)(bm + row) * MMD_DIMS + ks * MMD_KC + k4);
            f32x4 vb = *(const f32x4*)(B + (size_t)(bn + row) * MMD_DIMS + ks * MMD_KC + k4);
#pragma unroll
            for (int e = 0; e < 4; ++e) {
                __bf16 ha = mmd_bf_round(va[e]);
                sAhi[row * MMD_KPAD + k4 + e] = ha;
                sAlo[row * MMD_KPAD + k4 + e] = mmd_bf_round(va[e] - mmd_bf_to_f32(ha));
                __bf16 hb = mmd_bf_round(vb[e]);
                sBhi[row * MMD_KPAD + k4 + e] = hb;
                sBlo[row * MMD_KPAD + k4 + e] = mmd_bf_round(vb[e] - mmd_bf_to_f32(hb));
            }
        }
        if (ks + 1 < MMD_KSTEPS) {
            __builtin_prefetch(A + (size_t)(bm + (t >> 1)) * MMD_DIMS + (ks + 1) * MMD_KC, 0, 0);
            __builtin_prefetch(B + (size_t)(bn + (t >> 1)) * MMD_DIMS + (ks + 1) * MMD_KC, 0, 0);
        }
        __syncthreads();
        mmd_compute_step(sAhi, sAlo, sBhi, sBlo, lane, wm, wn, acc);
    }

    red[t] = mmd_epilogue(acc, ea, eb, bm, bn, wm, wn, lane);
    __syncthreads();
    for (int off = 128; off > 0; off >>= 1) { if (t < off) red[t] += red[t + off]; __syncthreads(); }
    if (t == 0)
        partials[(size_t)term * (MMD_TILES * MMD_TILES) + blockIdx.y * MMD_TILES + blockIdx.x] = red[0];
}

// ---------------------------------------------------------------------------
// Deterministic final reduction + sqrt
// ---------------------------------------------------------------------------
__global__ void mmd_finalize_kernel(const float* __restrict__ partials, float* __restrict__ out) {
    __shared__ float red[256];
    const int t = threadIdx.x;
    float sums[3];
    for (int term = 0; term < 3; ++term) {
        const float* p = partials + (size_t)term * (MMD_TILES * MMD_TILES);
        float s = 0.0f;
        for (int i = t; i < MMD_TILES * MMD_TILES; i += 256) s += p[i];
        red[t] = s; __syncthreads();
        for (int off = 128; off > 0; off >>= 1) { if (t < off) red[t] += red[t + off]; __syncthreads(); }
        sums[term] = red[0];
        __syncthreads();
    }
    if (t == 0) {
        float mmd = sums[0] + sums[1] - 2.0f * sums[2];
        out[0] = sqrtf(fmaxf(mmd, 0.0f));
    }
}

// ---------------------------------------------------------------------------
extern "C" void kernel_launch(void* const* d_in, const int* in_sizes, int n_in,
                              void* d_out, int out_size, void* d_ws, size_t ws_size,
                              hipStream_t stream) {
    const float* Nmat = (const float*)d_in[0];   // [8192,512] f32
    const float* Rmat = (const float*)d_in[1];   // [8192,512] f32
    const float* w    = (const float*)d_in[2];   // [8192]     f32

    float* ws       = (float*)d_ws;
    float* eNw      = ws;                        // 8192 f32
    float* eRu      = ws + MMD_NROWS;            // 8192 f32
    float* partials = ws + 2 * MMD_NROWS;        // 3 * 64*64 f32

    const size_t small_bytes = (size_t)(2 * MMD_NROWS + 3 * MMD_TILES * MMD_TILES) * sizeof(float);
    const size_t planes_off  = (small_bytes + 255) & ~(size_t)255;
    const size_t need        = planes_off + 4 * MMD_PLANE_ELEMS * sizeof(unsigned short);

    mmd_prep_kernel<<<MMD_NROWS, 128, 0, stream>>>(Nmat, Rmat, w, eNw, eRu);

    dim3 grid(MMD_TILES, MMD_TILES, 3);
    if (ws_size >= need) {
        unsigned short* Nhi = (unsigned short*)((char*)d_ws + planes_off);
        unsigned short* Nlo = Nhi + MMD_PLANE_ELEMS;
        unsigned short* Rhi = Nlo + MMD_PLANE_ELEMS;
        unsigned short* Rlo = Rhi + MMD_PLANE_ELEMS;
        mmd_convert_kernel<<<(int)(MMD_PLANE_ELEMS / (256 * 4)), 256, 0, stream>>>(
            Nmat, Rmat, Nhi, Nlo, Rhi, Rlo);
        mmd_wmma_async_kernel<<<grid, 256, 0, stream>>>(Nhi, Nlo, Rhi, Rlo, eNw, eRu, partials);
    } else {
        mmd_wmma_fb_kernel<<<grid, 256, 0, stream>>>(Nmat, Rmat, eNw, eRu, partials);
    }

    mmd_finalize_kernel<<<1, 256, 0, stream>>>(partials, (float*)d_out);
}
